// SparseMOE_39393440039229
// MI455X (gfx1250) — matrix operand order repaired and checked by reading, not compiled, not run on hardware
//
#include <hip/hip_runtime.h>
#include <math.h>

// Sparse MoE forward for gfx1250 (MI455X), wave32 + bf16 WMMA path.
// B=2, S=2048 -> NT=4096 tokens; d=1024; h=4096; E=8; top-k=2.

#define DM 1024
#define HD 4096
#define NE 8
#define NT 4096
#define CAP 4096          // worst-case tokens per expert
#define SLOTS (NT * 2)    // exact total routed slots (top-2, distinct experts)

typedef __attribute__((ext_vector_type(16))) __bf16 v16bf;
typedef __attribute__((ext_vector_type(8)))  float  v8f;

union BF16x16 { v16bf v; uint4 q[2]; };

__device__ __forceinline__ unsigned short f32_to_bf16(float f) {
  unsigned int u = __float_as_uint(f);
  u += 0x7fffu + ((u >> 16) & 1u);   // round-to-nearest-even
  return (unsigned short)(u >> 16);
}

// ---------------- utility kernels ----------------

__global__ void k_zero_out(float* __restrict__ out, int n) {
  int i = blockIdx.x * blockDim.x + threadIdx.x;
  int stride = gridDim.x * blockDim.x;
  for (; i < n; i += stride) out[i] = 0.f;
}

__global__ void k_zero_meta(int* __restrict__ count, int* __restrict__ cursor) {
  int t = threadIdx.x;
  if (t < NE) { count[t] = 0; cursor[t] = 0; }
}

__global__ void k_cvt_x(const float* __restrict__ x, unsigned short* __restrict__ xb, int n) {
  int i = blockIdx.x * blockDim.x + threadIdx.x;
  if (i < n) xb[i] = f32_to_bf16(x[i]);
}

// src [E][R][C] f32 -> dst [E][C][R] bf16 (tiled transpose via LDS)
__global__ void k_transpose_cvt(const float* __restrict__ src, unsigned short* __restrict__ dst,
                                int R, int C) {
  __shared__ float tile[32][33];
  const int e = blockIdx.z;
  const int c0 = blockIdx.x * 32, r0 = blockIdx.y * 32;
  const int tx = threadIdx.x, ty = threadIdx.y;
  const float* s = src + (size_t)e * R * C;
#pragma unroll
  for (int i = 0; i < 4; i++)
    tile[ty + 8 * i][tx] = s[(size_t)(r0 + ty + 8 * i) * C + (c0 + tx)];
  __syncthreads();
  unsigned short* d = dst + (size_t)e * C * R;
#pragma unroll
  for (int i = 0; i < 4; i++)
    d[(size_t)(c0 + ty + 8 * i) * R + (r0 + tx)] = f32_to_bf16(tile[tx][ty + 8 * i]);
}

// ---------------- router: noisy top-2 ----------------

__global__ void __launch_bounds__(128) k_router(
    const float* __restrict__ x, const float* __restrict__ noise,
    const float* __restrict__ Wr, const float* __restrict__ br,
    const float* __restrict__ Wn, const float* __restrict__ bn,
    int* __restrict__ topi, float* __restrict__ gates, int* __restrict__ count) {
  const int t = blockIdx.x * 4 + (threadIdx.x >> 5);   // one wave per token
  const int lane = threadIdx.x & 31;
  const float* xr = x + (size_t)t * DM;
  float lg[NE];
#pragma unroll
  for (int e = 0; e < NE; e++) {
    float ar = 0.f, an = 0.f;
    for (int i = lane; i < DM; i += 32) {
      float xv = xr[i];
      ar += xv * Wr[i * NE + e];
      an += xv * Wn[i * NE + e];
    }
#pragma unroll
    for (int off = 16; off > 0; off >>= 1) {
      ar += __shfl_xor(ar, off, 32);
      an += __shfl_xor(an, off, 32);
    }
    float z = an + bn[e];
    float sp = (z > 20.f) ? z : log1pf(expf(z));       // softplus
    lg[e] = ar + br[e] + sp * noise[(size_t)t * NE + e];
  }
  // top-2, first-occurrence tie-break (matches lax.top_k)
  int i0 = 0; float v0 = lg[0];
#pragma unroll
  for (int e = 1; e < NE; e++) if (lg[e] > v0) { v0 = lg[e]; i0 = e; }
  int i1 = (i0 == 0) ? 1 : 0; float v1 = lg[i1];
#pragma unroll
  for (int e = 0; e < NE; e++) if (e != i0 && lg[e] > v1) { v1 = lg[e]; i1 = e; }
  float g0 = 1.f / (1.f + expf(v1 - v0));              // softmax over {v0,v1}
  float g1 = 1.f - g0;
  if (lane == 0) {
    topi[t * 2 + 0] = i0; topi[t * 2 + 1] = i1;
    gates[t * 2 + 0] = g0; gates[t * 2 + 1] = g1;
    atomicAdd(&count[i0], 1);
    atomicAdd(&count[i1], 1);
  }
}

__global__ void k_prefix(const int* __restrict__ count, int* __restrict__ seg) {
  if (threadIdx.x == 0) {
    int s = 0;
    for (int e = 0; e < NE; e++) { seg[e] = s; s += count[e]; }
    seg[NE] = s;
  }
}

__global__ void k_scatter(const int* __restrict__ topi, const float* __restrict__ gates,
                          const int* __restrict__ seg, int* __restrict__ cursor,
                          int* __restrict__ rows, float* __restrict__ gslot) {
  int i = blockIdx.x * blockDim.x + threadIdx.x;       // 0..SLOTS-1
  if (i >= SLOTS) return;
  int e = topi[i];
  int pos = atomicAdd(&cursor[e], 1);
  int slot = seg[e] + pos;
  rows[slot] = i >> 1;                                 // token id
  gslot[slot] = gates[i];
}

// ---------------- expert GEMMs (bf16 WMMA, wave tile 32x64) ----------------
// A lane layout (16x32 bf16): elems 0..15 = K = lh*16+j ; row M = lane&15
// B lane layout (32x16 bf16): elems 0..15 = K = lh*16+j ; col N = lane&15  (contiguous in transposed weights)
// C/D (16x16 f32): VGPR v, lane -> (M = v + 8*lh, N = lane&15)
// Each wave: M=32 (2 A operands), N=64 (4 B operands) -> 8 WMMA / 6 b128-load-pairs per k-step.
// Block = 4 waves tiled in N -> block tile 32 x 256.

__global__ void __launch_bounds__(128) k_gemm1(
    const unsigned short* __restrict__ xb,    // [NT][DM] bf16
    const unsigned short* __restrict__ w1t,   // [NE][HD][DM] bf16 (transposed)
    const float* __restrict__ b1,             // [NE][HD]
    const int* __restrict__ rows,
    const int* __restrict__ seg, const int* __restrict__ count,
    unsigned short* __restrict__ hbuf) {      // [SLOTS][HD] bf16 (relu'd)
  const int e = blockIdx.z;
  const int Me = count[e];
  const int m0 = blockIdx.y * 32;
  if (m0 >= Me) return;
  const int segE = seg[e];
  const int lane = threadIdx.x & 31;
  const int w = threadIdx.x >> 5;              // 0..3, N-slice within block
  const int lm = lane & 15, lh = lane >> 4;
  int mr0 = m0 + lm;      if (mr0 > Me - 1) mr0 = Me - 1;   // clamp partial tile
  int mr1 = m0 + 16 + lm; if (mr1 > Me - 1) mr1 = Me - 1;
  const unsigned short* ap0 = xb + (size_t)rows[segE + mr0] * DM + lh * 16;
  const unsigned short* ap1 = xb + (size_t)rows[segE + mr1] * DM + lh * 16;
  const int nbase = blockIdx.x * 256 + w * 64;
  const unsigned short* bp0 = w1t + ((size_t)e * HD + (nbase + 0 * 16 + lm)) * DM + lh * 16;
  const unsigned short* bp1 = w1t + ((size_t)e * HD + (nbase + 1 * 16 + lm)) * DM + lh * 16;
  const unsigned short* bp2 = w1t + ((size_t)e * HD + (nbase + 2 * 16 + lm)) * DM + lh * 16;
  const unsigned short* bp3 = w1t + ((size_t)e * HD + (nbase + 3 * 16 + lm)) * DM + lh * 16;
  v8f a00 = {}, a01 = {}, a02 = {}, a03 = {};
  v8f a10 = {}, a11 = {}, a12 = {}, a13 = {};
  for (int k0 = 0; k0 < DM; k0 += 32) {
    BF16x16 a0, a1, b0, b1v, b2v, b3v;
    a0.q[0] = *(const uint4*)(ap0 + k0);
    a0.q[1] = *(const uint4*)(ap0 + k0 + 8);
    a1.q[0] = *(const uint4*)(ap1 + k0);
    a1.q[1] = *(const uint4*)(ap1 + k0 + 8);
    b0.q[0]  = *(const uint4*)(bp0 + k0); b0.q[1]  = *(const uint4*)(bp0 + k0 + 8);
    b1v.q[0] = *(const uint4*)(bp1 + k0); b1v.q[1] = *(const uint4*)(bp1 + k0 + 8);
    b2v.q[0] = *(const uint4*)(bp2 + k0); b2v.q[1] = *(const uint4*)(bp2 + k0 + 8);
    b3v.q[0] = *(const uint4*)(bp3 + k0); b3v.q[1] = *(const uint4*)(bp3 + k0 + 8);
    a00 = __builtin_amdgcn_wmma_f32_16x16x32_bf16(false, a0.v, false, b0.v,  (short)0, a00, false, false);
    a01 = __builtin_amdgcn_wmma_f32_16x16x32_bf16(false, a0.v, false, b1v.v, (short)0, a01, false, false);
    a02 = __builtin_amdgcn_wmma_f32_16x16x32_bf16(false, a0.v, false, b2v.v, (short)0, a02, false, false);
    a03 = __builtin_amdgcn_wmma_f32_16x16x32_bf16(false, a0.v, false, b3v.v, (short)0, a03, false, false);
    a10 = __builtin_amdgcn_wmma_f32_16x16x32_bf16(false, a1.v, false, b0.v,  (short)0, a10, false, false);
    a11 = __builtin_amdgcn_wmma_f32_16x16x32_bf16(false, a1.v, false, b1v.v, (short)0, a11, false, false);
    a12 = __builtin_amdgcn_wmma_f32_16x16x32_bf16(false, a1.v, false, b2v.v, (short)0, a12, false, false);
    a13 = __builtin_amdgcn_wmma_f32_16x16x32_bf16(false, a1.v, false, b3v.v, (short)0, a13, false, false);
  }
  v8f accs[2][4] = {{a00, a01, a02, a03}, {a10, a11, a12, a13}};
#pragma unroll
  for (int s = 0; s < 2; s++) {
#pragma unroll
    for (int j = 0; j < 4; j++) {
      int n = nbase + j * 16 + lm;
      float bias = b1[e * HD + n];
#pragma unroll
      for (int v = 0; v < 8; v++) {
        int mrow = m0 + s * 16 + v + 8 * lh;
        if (mrow < Me) {
          float val = accs[s][j][v] + bias;
          val = val > 0.f ? val : 0.f;               // relu fused here
          hbuf[(size_t)(segE + mrow) * HD + n] = f32_to_bf16(val);
        }
      }
    }
  }
}

__global__ void __launch_bounds__(128) k_gemm2(
    const unsigned short* __restrict__ hbuf,  // [SLOTS][HD] bf16
    const unsigned short* __restrict__ w2t,   // [NE][DM][HD] bf16 (transposed)
    const float* __restrict__ b2,             // [NE][DM]
    const int* __restrict__ rows, const float* __restrict__ gslot,
    const int* __restrict__ seg, const int* __restrict__ count,
    float* __restrict__ out) {                // [NT][DM] f32
  const int e = blockIdx.z;
  const int Me = count[e];
  const int m0 = blockIdx.y * 32;
  if (m0 >= Me) return;
  const int segE = seg[e];
  const int lane = threadIdx.x & 31;
  const int w = threadIdx.x >> 5;
  const int lm = lane & 15, lh = lane >> 4;
  int mr0 = m0 + lm;      if (mr0 > Me - 1) mr0 = Me - 1;
  int mr1 = m0 + 16 + lm; if (mr1 > Me - 1) mr1 = Me - 1;
  const unsigned short* ap0 = hbuf + (size_t)(segE + mr0) * HD + lh * 16;
  const unsigned short* ap1 = hbuf + (size_t)(segE + mr1) * HD + lh * 16;
  const int nbase = blockIdx.x * 256 + w * 64;
  const unsigned short* bp0 = w2t + ((size_t)e * DM + (nbase + 0 * 16 + lm)) * HD + lh * 16;
  const unsigned short* bp1 = w2t + ((size_t)e * DM + (nbase + 1 * 16 + lm)) * HD + lh * 16;
  const unsigned short* bp2 = w2t + ((size_t)e * DM + (nbase + 2 * 16 + lm)) * HD + lh * 16;
  const unsigned short* bp3 = w2t + ((size_t)e * DM + (nbase + 3 * 16 + lm)) * HD + lh * 16;
  v8f a00 = {}, a01 = {}, a02 = {}, a03 = {};
  v8f a10 = {}, a11 = {}, a12 = {}, a13 = {};
  for (int k0 = 0; k0 < HD; k0 += 32) {
    BF16x16 a0, a1, b0, b1v, b2v, b3v;
    a0.q[0] = *(const uint4*)(ap0 + k0);
    a0.q[1] = *(const uint4*)(ap0 + k0 + 8);
    a1.q[0] = *(const uint4*)(ap1 + k0);
    a1.q[1] = *(const uint4*)(ap1 + k0 + 8);
    b0.q[0]  = *(const uint4*)(bp0 + k0); b0.q[1]  = *(const uint4*)(bp0 + k0 + 8);
    b1v.q[0] = *(const uint4*)(bp1 + k0); b1v.q[1] = *(const uint4*)(bp1 + k0 + 8);
    b2v.q[0] = *(const uint4*)(bp2 + k0); b2v.q[1] = *(const uint4*)(bp2 + k0 + 8);
    b3v.q[0] = *(const uint4*)(bp3 + k0); b3v.q[1] = *(const uint4*)(bp3 + k0 + 8);
    a00 = __builtin_amdgcn_wmma_f32_16x16x32_bf16(false, a0.v, false, b0.v,  (short)0, a00, false, false);
    a01 = __builtin_amdgcn_wmma_f32_16x16x32_bf16(false, a0.v, false, b1v.v, (short)0, a01, false, false);
    a02 = __builtin_amdgcn_wmma_f32_16x16x32_bf16(false, a0.v, false, b2v.v, (short)0, a02, false, false);
    a03 = __builtin_amdgcn_wmma_f32_16x16x32_bf16(false, a0.v, false, b3v.v, (short)0, a03, false, false);
    a10 = __builtin_amdgcn_wmma_f32_16x16x32_bf16(false, a1.v, false, b0.v,  (short)0, a10, false, false);
    a11 = __builtin_amdgcn_wmma_f32_16x16x32_bf16(false, a1.v, false, b1v.v, (short)0, a11, false, false);
    a12 = __builtin_amdgcn_wmma_f32_16x16x32_bf16(false, a1.v, false, b2v.v, (short)0, a12, false, false);
    a13 = __builtin_amdgcn_wmma_f32_16x16x32_bf16(false, a1.v, false, b3v.v, (short)0, a13, false, false);
  }
  v8f accs[2][4] = {{a00, a01, a02, a03}, {a10, a11, a12, a13}};
#pragma unroll
  for (int s = 0; s < 2; s++) {
#pragma unroll
    for (int j = 0; j < 4; j++) {
      int n = nbase + j * 16 + lm;
      float bias = b2[e * DM + n];
#pragma unroll
      for (int v = 0; v < 8; v++) {
        int mrow = m0 + s * 16 + v + 8 * lh;
        if (mrow < Me) {
          int slot = segE + mrow;
          int tok = rows[slot];
          float g = gslot[slot];
          atomicAdd(out + (size_t)tok * DM + n, g * (accs[s][j][v] + bias));
        }
      }
    }
  }
}

// ---------------- launcher ----------------

extern "C" void kernel_launch(void* const* d_in, const int* in_sizes, int n_in,
                              void* d_out, int out_size, void* d_ws, size_t ws_size,
                              hipStream_t stream) {
  (void)in_sizes; (void)n_in; (void)ws_size;
  const float* x     = (const float*)d_in[0];
  const float* noise = (const float*)d_in[1];
  const float* Wr    = (const float*)d_in[2];
  const float* br    = (const float*)d_in[3];
  const float* Wn    = (const float*)d_in[4];
  const float* bn    = (const float*)d_in[5];
  const float* W1    = (const float*)d_in[6];
  const float* b1    = (const float*)d_in[7];
  const float* W2    = (const float*)d_in[8];
  const float* b2    = (const float*)d_in[9];
  float* out = (float*)d_out;

  char* ws = (char*)d_ws;
  int*   count  = (int*)(ws + 0);
  int*   cursor = (int*)(ws + 64);
  int*   seg    = (int*)(ws + 128);
  int*   topi   = (int*)(ws + 256);                          // NT*2 ints
  float* gates  = (float*)(ws + 256 + 32768);                // NT*2 f32
  int*   rows   = (int*)(ws + 256 + 65536);                  // SLOTS ints
  float* gslot  = (float*)(ws + 256 + 98304);                // SLOTS f32
  unsigned short* xb   = (unsigned short*)(ws + 256 + 131072);            // NT*DM bf16 (8 MiB)
  unsigned short* w1t  = xb  + (size_t)NT * DM;                           // E*HD*DM bf16 (64 MiB)
  unsigned short* w2t  = w1t + (size_t)NE * HD * DM;                      // E*DM*HD bf16 (64 MiB)
  unsigned short* hbuf = w2t + (size_t)NE * DM * HD;                      // SLOTS*HD bf16 (64 MiB)

  k_zero_out<<<2048, 256, 0, stream>>>(out, out_size);
  k_zero_meta<<<1, 32, 0, stream>>>(count, cursor);
  k_cvt_x<<<(NT * DM) / 256, 256, 0, stream>>>(x, xb, NT * DM);
  k_transpose_cvt<<<dim3(HD / 32, DM / 32, NE), dim3(32, 8), 0, stream>>>(W1, w1t, DM, HD);
  k_transpose_cvt<<<dim3(DM / 32, HD / 32, NE), dim3(32, 8), 0, stream>>>(W2, w2t, HD, DM);
  k_router<<<NT / 4, 128, 0, stream>>>(x, noise, Wr, br, Wn, bn, topi, gates, count);
  k_prefix<<<1, 32, 0, stream>>>(count, seg);
  k_scatter<<<SLOTS / 256, 256, 0, stream>>>(topi, gates, seg, cursor, rows, gslot);
  k_gemm1<<<dim3(HD / 256, CAP / 32, NE), 128, 0, stream>>>(xb, w1t, b1, rows, seg, count, hbuf);
  k_gemm2<<<dim3(DM / 256, CAP / 32, NE), 128, 0, stream>>>(hbuf, w2t, b2, rows, gslot, seg, count, out);
}
